// Mampa_60885456388760
// MI455X (gfx1250) — matrix operand order repaired and checked by reading, not compiled, not run on hardware
//
#include <hip/hip_runtime.h>
#include <hip/hip_bf16.h>

// ---------------------------------------------------------------------------
// Mamba selective-scan forward for MI455X (gfx1250), wave32.
// GEMMs: bf16 operands pre-packed in global memory, async global->LDS staging
// (ASYNCcnt, double buffered), v_wmma_f32_16x16x32_bf16 core, fp32 accum,
// transposed store out[n*ldo+m] so every downstream read is contiguous.
// ---------------------------------------------------------------------------

typedef __attribute__((ext_vector_type(8)))  float   v8f;
typedef __attribute__((ext_vector_type(8)))  __bf16  v8bf;
typedef __attribute__((ext_vector_type(16))) __bf16  v16bf;

static constexpr int Bn  = 4;
static constexpr int L   = 1024;
static constexpr int DM  = 1024;
static constexpr int DI  = 2048;
static constexpr int E2  = 2 * DI;     // 4096
static constexpr int NST = 16;         // d_state
static constexpr int XDB = 96;         // dt_rank + 2*d_state

__device__ __forceinline__ unsigned short f2bf_u(float f) {
    union { float f; unsigned u; } v; v.f = f;
    unsigned u = v.u + 0x7FFFu + ((v.u >> 16) & 1u);   // round-to-nearest-even
    return (unsigned short)(u >> 16);
}
__device__ __forceinline__ unsigned pack2(float x, float y) {
    return (unsigned)f2bf_u(x) | ((unsigned)f2bf_u(y) << 16);
}

__device__ __forceinline__ v16bf cat8(const __bf16* p0, const __bf16* p1) {
    v8bf a = *(const v8bf*)p0;
    v8bf b = *(const v8bf*)p1;
    return __builtin_shufflevector(a, b, 0,1,2,3,4,5,6,7,8,9,10,11,12,13,14,15);
}

// CDNA5 async global->LDS copy (16B per lane), tracked by ASYNCcnt.
__device__ __forceinline__ void async_g2l_b128(unsigned lds_off, const void* gptr) {
    asm volatile("global_load_async_to_lds_b128 %0, %1, off"
                 :: "v"(lds_off), "v"(gptr) : "memory");
}
__device__ __forceinline__ void wait_async4() {
    asm volatile("s_wait_asynccnt 0x4" ::: "memory");
}
__device__ __forceinline__ void wait_async0() {
    asm volatile("s_wait_asynccnt 0x0" ::: "memory");
}

// ---------------------------------------------------------------------------
// Tiled WMMA GEMM, bf16 operands. Block tile 128(M) x 128(N), 8 waves.
// Wave w owns 32x64: wm = w&3 (M), wn = w>>2 (N) -> 2x4 subtiles of 16x16.
// EPI: 0 = identity, 1 = softplus. OUT2: also store bf16 copy of output.
// ---------------------------------------------------------------------------
template <int EPI, int OUT2>
__global__ __launch_bounds__(256)
void wmma_gemm_tn(const __bf16* __restrict__ A,   // M x K, row stride lda
                  const __bf16* __restrict__ B,   // (per batch) N x K, row stride ldb
                  float*        __restrict__ O,   // (per batch) O[n*ldo + m]
                  __bf16*       __restrict__ O2,  // optional bf16 copy, same layout
                  int M, int K, int lda, int ldb, int ldo,
                  long strideB, long strideO)
{
    const int tid  = threadIdx.x;
    const int lane = tid & 31;
    const int wave = tid >> 5;
    const int wm   = wave & 3;
    const int wn   = wave >> 2;
    const int blockM = blockIdx.x * 128;
    const int blockN = blockIdx.y * 128;

    const __bf16* Bb = B + (long)blockIdx.z * strideB;
    float*        Ob = O + (long)blockIdx.z * strideO;

    // [buf][A/B][128 rows x 40 bf16] ; 80B row pitch -> conflict-free b128 reads
    __shared__ __align__(16) __bf16 lds[2][2][128 * 40];
    const unsigned baseA[2] = { (unsigned)(unsigned long long)&lds[0][0][0],
                                (unsigned)(unsigned long long)&lds[1][0][0] };
    const unsigned baseB[2] = { (unsigned)(unsigned long long)&lds[0][1][0],
                                (unsigned)(unsigned long long)&lds[1][1][0] };

    // stage one K-slab (A 128x32 + B 128x32 bf16): 4 async b128 ops per thread
    auto stage = [&](int buf, int k0) {
#pragma unroll
        for (int i = 0; i < 2; ++i) {
            int slot = tid + i * 256;          // 0..511
            int row  = slot >> 2;
            int cq   = slot & 3;               // 16B chunk within 64B row
            int ga   = blockM + row;
            if (ga >= M) ga = M - 1;           // clamp: keep 4 async ops/thread
            async_g2l_b128(baseA[buf] + (unsigned)(row * 80 + cq * 16),
                           (const char*)A + ((long)ga * lda + k0) * 2 + cq * 16);
            async_g2l_b128(baseB[buf] + (unsigned)(row * 80 + cq * 16),
                           (const char*)Bb + ((long)(blockN + row) * ldb + k0) * 2 + cq * 16);
        }
    };

    v8f acc[2][4] = {};
    const int KT = K / 32;

    stage(0, 0);
    for (int kt = 0; kt < KT; ++kt) {
        if (kt + 1 < KT) { stage((kt + 1) & 1, (kt + 1) * 32); wait_async4(); }
        else             { wait_async0(); }
        __syncthreads();                       // stage kt visible to all waves

        const __bf16* Al = &lds[kt & 1][0][0];
        const __bf16* Bl = &lds[kt & 1][1][0];

        // A 16x32 frag: lane=row M; K chunks {0,16} (lanes<16) / {8,24}
        const int arow = wm * 32 + (lane & 15);
        const int ac0  = (lane < 16) ? 0 : 8;
        v16bf afrag[2];
#pragma unroll
        for (int mi = 0; mi < 2; ++mi) {
            const __bf16* base = Al + (arow + mi * 16) * 40;
            afrag[mi] = cat8(base + ac0, base + ac0 + 16);
        }
        // B 32x16 frag: lane=col N; K chunks {0,16} (lanes<16) / {8,24}
        const int brow = wn * 64 + (lane & 15);
        const int bc0  = (lane < 16) ? 0 : 8;
        v16bf bfrag[4];
#pragma unroll
        for (int ni = 0; ni < 4; ++ni) {
            const __bf16* base = Bl + (brow + ni * 16) * 40;
            bfrag[ni] = cat8(base + bc0, base + bc0 + 16);
        }

#pragma unroll
        for (int mi = 0; mi < 2; ++mi)
#pragma unroll
            for (int ni = 0; ni < 4; ++ni)
                acc[mi][ni] = __builtin_amdgcn_wmma_f32_16x16x32_bf16(
                    false, afrag[mi], false, bfrag[ni],
                    (short)0, acc[mi][ni], false, false);
        __syncthreads();                       // protect buf before re-staging
    }

    // transposed store: D lane holds 8 contiguous M values at column N
#pragma unroll
    for (int mi = 0; mi < 2; ++mi) {
        int mg = blockM + wm * 32 + mi * 16;
        if (mg >= M) continue;
        int m = mg + ((lane < 16) ? 0 : 8);
#pragma unroll
        for (int ni = 0; ni < 4; ++ni) {
            int n = blockN + wn * 64 + ni * 16 + (lane & 15);
            v8f a = acc[mi][ni];
            if (EPI == 1) {
#pragma unroll
                for (int e = 0; e < 8; ++e) {
                    float x = a[e];
                    a[e] = (x > 20.f) ? x : log1pf(__expf(x));
                }
            }
            float* p = Ob + (long)n * ldo + m;
            *(float4*)(p)     = make_float4(a[0], a[1], a[2], a[3]);
            *(float4*)(p + 4) = make_float4(a[4], a[5], a[6], a[7]);
            if (OUT2) {
                __bf16* p2 = O2 + (long)blockIdx.z * strideO + (long)n * ldo + m;
                uint4 u;
                u.x = pack2(a[0], a[1]); u.y = pack2(a[2], a[3]);
                u.z = pack2(a[4], a[5]); u.w = pack2(a[6], a[7]);
                *(uint4*)p2 = u;
            }
        }
    }
}

// ---------------------------------------------------------------------------
// fp32 -> bf16 packer (float4 -> 4 bf16 per thread).
// ---------------------------------------------------------------------------
__global__ __launch_bounds__(256)
void pack_bf16_kernel(const float* __restrict__ src, __bf16* __restrict__ dst, long n4)
{
    long i = (long)blockIdx.x * blockDim.x + threadIdx.x;
    if (i >= n4) return;
    float4 v = ((const float4*)src)[i];
    ((uint2*)dst)[i] = make_uint2(pack2(v.x, v.y), pack2(v.z, v.w));
}

// ---------------------------------------------------------------------------
// Depthwise causal conv(k=4) + bias + SiLU; writes fp32 and bf16 copies.
// ---------------------------------------------------------------------------
__global__ __launch_bounds__(256)
void conv_silu_kernel(const float* __restrict__ xz,   // (b,l,4096)
                      const float* __restrict__ cw,   // (di,4)
                      const float* __restrict__ cb,   // (di)
                      float* __restrict__ xc,         // (b,l,2048)
                      __bf16* __restrict__ xc_bf)
{
    long idx = (long)blockIdx.x * blockDim.x + threadIdx.x;
    if (idx >= (long)Bn * L * DI) return;
    int d = (int)(idx % DI);
    long t = idx / DI;
    int l = (int)(t % L);
    long bl0 = (t / L) * L;
    float acc = cb[d];
#pragma unroll
    for (int k = 0; k < 4; ++k) {
        int ls = l - 3 + k;
        if (ls >= 0)
            acc += cw[d * 4 + k] * xz[(bl0 + ls) * (long)E2 + d];
    }
    float s = acc * (1.f / (1.f + __expf(-acc)));
    long o = (bl0 + l) * (long)DI + d;
    xc[o] = s;
    union { unsigned short u; __bf16 h; } cvt; cvt.u = f2bf_u(s);
    xc_bf[o] = cvt.h;
}

// ---------------------------------------------------------------------------
// Selective scan: one lane per (b,d,n); y = sum_n h_n*C_n via shfl_xor (wave32).
// ---------------------------------------------------------------------------
__global__ __launch_bounds__(256)
void scan_kernel(const float* __restrict__ delta,  // (b,l,di)
                 const float* __restrict__ xdbl,   // (b,l,96)
                 const float* __restrict__ xc,     // (b,l,di)
                 const float* __restrict__ A_log,  // (di,16)
                 float* __restrict__ y)            // (b,l,di)
{
    int tid = blockIdx.x * blockDim.x + threadIdx.x;
    int n = tid & (NST - 1);
    int c = tid >> 4;
    if (c >= Bn * DI) return;
    int d = c % DI;
    long bl0 = (long)(c / DI) * L;
    float Av = -__expf(A_log[d * NST + n]);
    float h = 0.f;
    for (int l = 0; l < L; ++l) {
        long r = bl0 + l;
        float dl = delta[r * DI + d];
        float u  = xc[r * DI + d];
        const float* xd = xdbl + r * XDB;
        float Bv = xd[64 + n];
        float Cv = xd[80 + n];
        h = __expf(dl * Av) * h + (dl * Bv) * u;
        float p = h * Cv;
        p += __shfl_xor(p, 1);
        p += __shfl_xor(p, 2);
        p += __shfl_xor(p, 4);
        p += __shfl_xor(p, 8);
        if (n == 0) y[r * DI + d] = p;
    }
}

// ---------------------------------------------------------------------------
// od_bf = bf16( (y + xc*D) * silu(z) )
// ---------------------------------------------------------------------------
__global__ __launch_bounds__(256)
void gate_kernel(const float* __restrict__ y,
                 const float* __restrict__ xc,
                 const float* __restrict__ xz,
                 const float* __restrict__ Dv,
                 __bf16* __restrict__ od_bf)
{
    long idx = (long)blockIdx.x * blockDim.x + threadIdx.x;
    if (idx >= (long)Bn * L * DI) return;
    int d = (int)(idx % DI);
    long r = idx / DI;
    float z = xz[r * (long)E2 + DI + d];
    float sz = z * (1.f / (1.f + __expf(-z)));
    float v = (y[idx] + xc[idx] * Dv[d]) * sz;
    union { unsigned short u; __bf16 h; } cvt; cvt.u = f2bf_u(v);
    od_bf[idx] = cvt.h;
}

// ---------------------------------------------------------------------------
extern "C" void kernel_launch(void* const* d_in, const int* in_sizes, int n_in,
                              void* d_out, int out_size, void* d_ws, size_t ws_size,
                              hipStream_t stream) {
    (void)in_sizes; (void)n_in; (void)out_size; (void)ws_size;
    const float* hs    = (const float*)d_in[0];
    const float* w_in  = (const float*)d_in[1];
    const float* cw    = (const float*)d_in[2];
    const float* cb    = (const float*)d_in[3];
    const float* w_xp  = (const float*)d_in[4];
    const float* w_dt  = (const float*)d_in[5];
    const float* A_log = (const float*)d_in[6];
    const float* Dv    = (const float*)d_in[7];
    const float* w_out = (const float*)d_in[8];
    float* out = (float*)d_out;

    // fp32 workspace
    float* ws   = (float*)d_ws;
    float* xz   = ws;                            // (b,l,4096)
    float* xc   = xz   + (long)Bn * L * E2;      // (b,l,2048)
    float* xdbl = xc   + (long)Bn * L * DI;      // (b,l,96)
    float* dlt  = xdbl + (long)Bn * L * XDB;     // (b,l,2048)
    float* yb   = dlt  + (long)Bn * L * DI;      // (b,l,2048)
    // bf16 workspace
    __bf16* hs_bf    = (__bf16*)(yb + (long)Bn * L * DI);
    __bf16* w_in_bf  = hs_bf    + (long)Bn * L * DM;
    __bf16* w_xp_bf  = w_in_bf  + (long)E2 * DM;
    __bf16* w_dt_bf  = w_xp_bf  + (long)XDB * DI;
    __bf16* w_out_bf = w_dt_bf  + (long)DI * 64;
    __bf16* xc_bf    = w_out_bf + (long)DM * DI;
    __bf16* xdbl_bf  = xc_bf    + (long)Bn * L * DI;
    __bf16* od_bf    = xdbl_bf  + (long)Bn * L * XDB;

    auto pack = [&](const float* s, __bf16* dst, long n) {
        long n4 = n / 4;
        pack_bf16_kernel<<<(int)((n4 + 255) / 256), 256, 0, stream>>>(s, dst, n4);
    };
    pack(hs,    hs_bf,    (long)Bn * L * DM);
    pack(w_in,  w_in_bf,  (long)E2 * DM);
    pack(w_xp,  w_xp_bf,  (long)XDB * DI);
    pack(w_dt,  w_dt_bf,  (long)DI * 64);
    pack(w_out, w_out_bf, (long)DM * DI);

    // 1) in_proj: xz[b,l,e] = sum_d hs[b,l,d] * w_in[e,d]
    wmma_gemm_tn<0,0><<<dim3(E2 / 128, L / 128, Bn), 256, 0, stream>>>(
        w_in_bf, hs_bf, xz, nullptr, E2, DM, DM, DM, E2,
        (long)L * DM, (long)L * E2);

    // 2) depthwise conv + bias + silu (fp32 + bf16 copies)
    conv_silu_kernel<<<(Bn * L * DI) / 256, 256, 0, stream>>>(xz, cw, cb, xc, xc_bf);

    // 3) x_proj (M=96): fp32 for scan + bf16 for dt_proj
    wmma_gemm_tn<0,1><<<dim3(1, L / 128, Bn), 256, 0, stream>>>(
        w_xp_bf, xc_bf, xdbl, xdbl_bf, XDB, DI, DI, DI, XDB,
        (long)L * DI, (long)L * XDB);

    // 4) dt_proj + softplus
    wmma_gemm_tn<1,0><<<dim3(DI / 128, L / 128, Bn), 256, 0, stream>>>(
        w_dt_bf, xdbl_bf, dlt, nullptr, DI, 64, 64, XDB, DI,
        (long)L * XDB, (long)L * DI);

    // 5) selective scan
    scan_kernel<<<(Bn * DI * NST) / 256, 256, 0, stream>>>(dlt, xdbl, xc, A_log, yb);

    // 6) gate -> bf16
    gate_kernel<<<(Bn * L * DI) / 256, 256, 0, stream>>>(yb, xc, xz, Dv, od_bf);

    // 7) out_proj -> d_out (fp32)
    wmma_gemm_tn<0,0><<<dim3(DM / 128, L / 128, Bn), 256, 0, stream>>>(
        w_out_bf, od_bf, out, nullptr, DM, DI, DI, DI, DM,
        (long)L * DI, (long)L * DM);
}
